// Model_2_10995116278168
// MI455X (gfx1250) — compile-verified, matrix operand order checked by reading, not run on hardware
//
#include <hip/hip_runtime.h>
#include <math.h>

// ---------------------------------------------------------------------------
// Model: dual-branch GATv2 + TopKPooling GNN (B=64 graphs, 512 nodes, 8192
// edges each).  Dense node transforms run on v_wmma_f32_16x16x32_bf16; the
// irregular edge softmax/aggregation runs wave-per-edge (wave32, float4 per
// lane = 128 channels/row) with float atomics.
// ---------------------------------------------------------------------------

typedef __attribute__((ext_vector_type(16))) __bf16 v16bf;
typedef __attribute__((ext_vector_type(8)))  float  v8f;

#define B_    64
#define NPER  512
#define EPER  8192
#define FIN   64
#define C_    128
#define N1    (B_*NPER)   // 32768
#define E_    (B_*EPER)   // 524288
#define K1_   (NPER/2)    // 256
#define K2_   (NPER/4)    // 128
#define N2    (B_*K1_)    // 16384
#define N3    (B_*K2_)    // 8192
#define NEG_SLOPE   0.2f
#define OUTER_SLOPE 0.01f
#define NEGINF_ENC  0x007FFFFFu   // fenc(-inf)

static inline int cdiv(int a, int b) { return (a + b - 1) / b; }

// ---- helpers ---------------------------------------------------------------
__device__ __forceinline__ unsigned fenc(float f) {           // order-preserving
  unsigned u = __float_as_uint(f);
  return (u & 0x80000000u) ? ~u : (u | 0x80000000u);
}
__device__ __forceinline__ float fdec(unsigned u) {
  unsigned v = (u & 0x80000000u) ? (u & 0x7fffffffu) : ~u;
  return __uint_as_float(v);
}
__device__ __forceinline__ unsigned short f2bf(float f) {     // fp32 -> bf16 RNE
  unsigned u = __float_as_uint(f);
  u += 0x7fffu + ((u >> 16) & 1u);
  return (unsigned short)(u >> 16);
}

// ---- generic fills / converts ---------------------------------------------
__global__ void fill_f32_k(float* p, float v, int n) {
  int i = blockIdx.x * blockDim.x + threadIdx.x;
  if (i < n) p[i] = v;
}
__global__ void fill_u32_k(unsigned* p, unsigned v, int n) {
  int i = blockIdx.x * blockDim.x + threadIdx.x;
  if (i < n) p[i] = v;
}
__global__ void cvt_bf16_k(const float* __restrict__ in, unsigned short* __restrict__ out, int n) {
  int i = blockIdx.x * blockDim.x + threadIdx.x;
  if (i < n) out[i] = f2bf(in[i]);
}
// [K,N] fp32 row-major -> [N,K] bf16 row-major (transposed weights for WMMA B)
__global__ void cvt_bf16_t_k(const float* __restrict__ in, unsigned short* __restrict__ out,
                             int K, int N) {
  int i = blockIdx.x * blockDim.x + threadIdx.x;
  if (i >= K * N) return;
  int k = i / N, n = i % N;
  out[(size_t)n * K + k] = f2bf(in[i]);
}

// ---- WMMA GEMM: C[M,N] = A[M,K] * B[K,N], A bf16 row-major, B pre-transposed
// to Bt[N,K] bf16 row-major.  One 16x16 C tile per wave, blockDim=(32,8).
__global__ void gemm_bf16_wmma_k(const unsigned short* __restrict__ A,
                                 const unsigned short* __restrict__ Bt,
                                 float* __restrict__ C, int M, int N, int K) {
  int lane  = threadIdx.x;                       // 0..31
  int tile  = blockIdx.x * blockDim.y + threadIdx.y;
  int tilesN = N >> 4;
  int tiles  = (M >> 4) * tilesN;
  if (tile >= tiles) return;
  int tm = (tile / tilesN) << 4;
  int tn = (tile % tilesN) << 4;
  int half = lane >> 4, lm = lane & 15;
  const unsigned short* Arow = A  + (size_t)(tm + lm) * K;
  const unsigned short* Brow = Bt + (size_t)(tn + lm) * K;
  v8f acc = {};
  for (int k0 = 0; k0 < K; k0 += 32) {
    union { v16bf v; unsigned u[8]; } a, b;
#pragma unroll
    for (int r = 0; r < 8; ++r) {
      // ISA 16-bit A layout: VGPR r holds K pair at this offset (per lane half)
      int k = k0 + ((r < 4) ? (half * 8 + r * 2) : (16 + half * 8 + (r - 4) * 2));
      a.u[r] = *(const unsigned*)(Arow + k);
      b.u[r] = *(const unsigned*)(Brow + k);
    }
    acc = __builtin_amdgcn_wmma_f32_16x16x32_bf16(false, a.v, false, b.v,
                                                  (short)0, acc, false, false);
  }
  int mb = half * 8;                             // C layout: VGPR r -> row r+8*half
#pragma unroll
  for (int r = 0; r < 8; ++r)
    C[(size_t)(tm + mb + r) * N + tn + lm] = acc[r];
}

// ---- GATv2 edge score: e = mask ? leaky_relu(xl[s]+xr[d],0.2) . att : -1e30
// wave-per-edge; edges [0,E) from arrays, [E,E+n) are self loops.
__global__ void edge_score_k(const float* __restrict__ xl, const float* __restrict__ xr,
                             const int* __restrict__ src, const int* __restrict__ dst,
                             const int* __restrict__ mask, const float* __restrict__ att,
                             float* __restrict__ esc, int E, int n) {
  int gt = blockIdx.x * blockDim.x + threadIdx.x;
  int gw = gt >> 5, lane = gt & 31;
  if (gw >= E + n) return;
  int s, d, valid;
  if (gw < E) { s = src[gw]; d = dst[gw]; valid = mask ? mask[gw] : 1; }
  else        { s = gw - E;  d = s;       valid = 1; }
  float r = 0.f;
  if (valid) {
    float4 a = ((const float4*)(xl + (size_t)s * C_))[lane];
    float4 b = ((const float4*)(xr + (size_t)d * C_))[lane];
    float4 w = ((const float4*)att)[lane];
    float h;
    h = a.x + b.x; h = (h > 0.f) ? h : NEG_SLOPE * h; r += h * w.x;
    h = a.y + b.y; h = (h > 0.f) ? h : NEG_SLOPE * h; r += h * w.y;
    h = a.z + b.z; h = (h > 0.f) ? h : NEG_SLOPE * h; r += h * w.z;
    h = a.w + b.w; h = (h > 0.f) ? h : NEG_SLOPE * h; r += h * w.w;
  }
  for (int off = 16; off; off >>= 1) r += __shfl_down(r, off, 32);
  if (lane == 0) esc[gw] = valid ? r : -1e30f;
}

__global__ void seg_max_k(const float* __restrict__ esc, const int* __restrict__ dst,
                          int E, int n, unsigned* __restrict__ emax) {
  int e = blockIdx.x * blockDim.x + threadIdx.x;
  if (e >= E + n) return;
  float v = esc[e];
  if (v <= -1e29f) return;                     // masked edge; self-loops cover every node
  int d = (e < E) ? dst[e] : e - E;
  atomicMax(emax + d, fenc(v));
}

__global__ void dec_max_k(unsigned* emax, int n) {
  int i = blockIdx.x * blockDim.x + threadIdx.x;
  if (i < n) { unsigned u = emax[i]; ((float*)emax)[i] = fdec(u); }
}

__global__ void exp_sum_k(float* __restrict__ esc, const int* __restrict__ dst,
                          const float* __restrict__ emaxf, float* __restrict__ denom,
                          int E, int n) {
  int e = blockIdx.x * blockDim.x + threadIdx.x;
  if (e >= E + n) return;
  float v = esc[e];
  int d = (e < E) ? dst[e] : e - E;
  float ee = (v <= -1e29f) ? 0.f : expf(v - emaxf[d]);
  esc[e] = ee;
  if (ee != 0.f) atomicAdd(denom + d, ee);
}

__global__ void aggregate_k(const float* __restrict__ esc, const int* __restrict__ src,
                            const int* __restrict__ dst, const float* __restrict__ denom,
                            const float* __restrict__ xl, float* __restrict__ out,
                            int E, int n) {
  int gt = blockIdx.x * blockDim.x + threadIdx.x;
  int gw = gt >> 5, lane = gt & 31;
  if (gw >= E + n) return;
  float ee = esc[gw];
  if (ee == 0.f) return;
  int s = (gw < E) ? src[gw] : gw - E;
  int d = (gw < E) ? dst[gw] : gw - E;
  float alpha = ee / denom[d];
  float4 a = ((const float4*)(xl + (size_t)s * C_))[lane];
  float* o = out + (size_t)d * C_ + lane * 4;
  atomicAdd(o + 0, alpha * a.x);
  atomicAdd(o + 1, alpha * a.y);
  atomicAdd(o + 2, alpha * a.z);
  atomicAdd(o + 3, alpha * a.w);
}

__global__ void bias_lrelu_k(float* __restrict__ x, const float* __restrict__ bias, int total) {
  int i = blockIdx.x * blockDim.x + threadIdx.x;
  if (i >= total) return;
  float v = x[i] + bias[i & (C_ - 1)];
  x[i] = (v > 0.f) ? v : OUTER_SLOPE * v;
}

// ---- TopK pooling ----------------------------------------------------------
__global__ void invnorm_k(const float* __restrict__ w, float* __restrict__ out) {
  __shared__ float sh[C_];
  int t = threadIdx.x;
  sh[t] = w[t] * w[t];
  __syncthreads();
  for (int s = C_ / 2; s; s >>= 1) { if (t < s) sh[t] += sh[t + s]; __syncthreads(); }
  if (t == 0) out[0] = rsqrtf(sh[0]);
}

__global__ void node_score_k(const float* __restrict__ x, const float* __restrict__ w,
                             const float* __restrict__ invn, float* __restrict__ score, int n) {
  int gt = blockIdx.x * blockDim.x + threadIdx.x;
  int gw = gt >> 5, lane = gt & 31;
  if (gw >= n) return;
  float4 a = ((const float4*)(x + (size_t)gw * C_))[lane];
  float4 b = ((const float4*)w)[lane];
  float r = a.x * b.x + a.y * b.y + a.z * b.z + a.w * b.w;
  for (int off = 16; off; off >>= 1) r += __shfl_down(r, off, 32);
  if (lane == 0) score[gw] = tanhf(r * invn[0]);
}

// rank-select top-k per graph (matches lax.top_k ordering incl. tie-break)
__global__ void pool_k(const float* __restrict__ x, const float* __restrict__ score,
                       float* __restrict__ xnew, int* __restrict__ newid,
                       int n_per, int k) {
  __shared__ float sh[NPER];
  int b = blockIdx.x, i = threadIdx.x;       // blockDim.x == n_per
  int gi = b * n_per + i;
  float si = score[gi];
  sh[i] = si;
  __syncthreads();
  int rank = 0;
  for (int j = 0; j < n_per; ++j) {
    float sj = sh[j];
    rank += (sj > si) || (sj == si && j < i);
  }
  int nid = (rank < k) ? (b * k + rank) : -1;
  newid[gi] = nid;
  if (nid >= 0) {
    const float* xs = x + (size_t)gi * C_;
    float* xd = xnew + (size_t)nid * C_;
    for (int c = 0; c < C_; ++c) xd[c] = xs[c] * si;
  }
}

__global__ void remap_k(const int* __restrict__ src, const int* __restrict__ dst,
                        const int* __restrict__ mask, const int* __restrict__ newid,
                        int* __restrict__ src2, int* __restrict__ dst2,
                        int* __restrict__ mask2, int E) {
  int e = blockIdx.x * blockDim.x + threadIdx.x;
  if (e >= E) return;
  int s = newid[src[e]], d = newid[dst[e]];
  int m = (mask ? mask[e] : 1) && (s >= 0) && (d >= 0);
  src2[e] = (s < 0) ? 0 : s;
  dst2[e] = (d < 0) ? 0 : d;
  mask2[e] = m;
}

// [max, mean] readout accumulated into g[b, colOff .. colOff+255]
__global__ void readout_k(const float* __restrict__ x, float* __restrict__ g,
                          int k, int colOff) {
  int b = blockIdx.x, c = threadIdx.x;       // 128 threads
  const float* xb = x + (size_t)b * k * C_;
  float mx = -3.4e38f, sm = 0.f;
  for (int j = 0; j < k; ++j) {
    float v = xb[(size_t)j * C_ + c];
    mx = fmaxf(mx, v);
    sm += v;
  }
  g[b * 512 + colOff + c]        += mx;
  g[b * 512 + colOff + C_ + c]   += sm / (float)k;
}

// ---- tiny fp32 head --------------------------------------------------------
__global__ void linear_head_k(const float* __restrict__ X, const float* __restrict__ W,
                              const float* __restrict__ bias, float* __restrict__ Y,
                              int M, int K, int N, int act) {
  int i = blockIdx.x * blockDim.x + threadIdx.x;
  if (i >= M * N) return;
  int m = i / N, n = i % N;
  float acc = bias[n];
  for (int k = 0; k < K; ++k) acc += X[m * K + k] * W[k * N + n];
  if (act) acc = (acc > 0.f) ? acc : OUTER_SLOPE * acc;
  Y[i] = acc;
}

__global__ void logsoftmax2_k(const float* __restrict__ X, float* __restrict__ out, int M) {
  int m = blockIdx.x * blockDim.x + threadIdx.x;
  if (m >= M) return;
  float a = X[2 * m], b = X[2 * m + 1];
  float mx = fmaxf(a, b);
  float lse = mx + logf(expf(a - mx) + expf(b - mx));
  out[2 * m]     = a - lse;
  out[2 * m + 1] = b - lse;
}

// ---------------------------------------------------------------------------
struct Scratch {
  unsigned short *bfA, *bfWl, *bfWr;
  float *xl, *xr, *xagg, *xnew;
  float *esc; unsigned *emax; float *denom, *score; int *newid;
  int *srcB, *dstB, *maskB;
  float *invn, *gcat, *h1, *h2, *h3;
};

static void gat_tail(const int* src, const int* dst, const int* mask,
                     const float* att, const float* bias, int n,
                     const Scratch& S, hipStream_t st) {
  const int TB = 256;
  int ET = E_ + n;
  fill_u32_k<<<cdiv(n, TB), TB, 0, st>>>(S.emax, NEGINF_ENC, n);
  fill_f32_k<<<cdiv(n, TB), TB, 0, st>>>(S.denom, 0.f, n);
  fill_f32_k<<<cdiv(n * C_, TB), TB, 0, st>>>(S.xagg, 0.f, n * C_);
  edge_score_k<<<cdiv(ET * 32, TB), TB, 0, st>>>(S.xl, S.xr, src, dst, mask, att, S.esc, E_, n);
  seg_max_k<<<cdiv(ET, TB), TB, 0, st>>>(S.esc, dst, E_, n, S.emax);
  dec_max_k<<<cdiv(n, TB), TB, 0, st>>>(S.emax, n);
  exp_sum_k<<<cdiv(ET, TB), TB, 0, st>>>(S.esc, dst, (const float*)S.emax, S.denom, E_, n);
  aggregate_k<<<cdiv(ET * 32, TB), TB, 0, st>>>(S.esc, src, dst, S.denom, S.xl, S.xagg, E_, n);
  bias_lrelu_k<<<cdiv(n * C_, TB), TB, 0, st>>>(S.xagg, bias, n * C_);
}

static void run_branch(const float* x_in, const int* src0, const int* dst0,
                       const float* w1l, const float* w1r, const float* w1a,
                       const float* w1b, const float* p1w,
                       const float* w2l, const float* w2r, const float* w2a,
                       const float* w2b, const float* p2w,
                       int colOff, const Scratch& S, hipStream_t st) {
  const int TB = 256;
  dim3 gblk(32, 8);
  // ---- GATv2 layer 1 (K = FIN) ----
  cvt_bf16_k<<<cdiv(N1 * FIN, TB), TB, 0, st>>>(x_in, S.bfA, N1 * FIN);
  cvt_bf16_t_k<<<cdiv(FIN * C_, TB), TB, 0, st>>>(w1l, S.bfWl, FIN, C_);
  cvt_bf16_t_k<<<cdiv(FIN * C_, TB), TB, 0, st>>>(w1r, S.bfWr, FIN, C_);
  int tiles1 = (N1 / 16) * (C_ / 16);
  gemm_bf16_wmma_k<<<cdiv(tiles1, 8), gblk, 0, st>>>(S.bfA, S.bfWl, S.xl, N1, C_, FIN);
  gemm_bf16_wmma_k<<<cdiv(tiles1, 8), gblk, 0, st>>>(S.bfA, S.bfWr, S.xr, N1, C_, FIN);
  gat_tail(src0, dst0, nullptr, w1a, w1b, N1, S, st);
  // ---- pool 1: 512 -> 256 per graph ----
  invnorm_k<<<1, C_, 0, st>>>(p1w, S.invn);
  node_score_k<<<cdiv(N1 * 32, TB), TB, 0, st>>>(S.xagg, p1w, S.invn, S.score, N1);
  pool_k<<<B_, NPER, 0, st>>>(S.xagg, S.score, S.xnew, S.newid, NPER, K1_);
  remap_k<<<cdiv(E_, TB), TB, 0, st>>>(src0, dst0, nullptr, S.newid, S.srcB, S.dstB, S.maskB, E_);
  readout_k<<<B_, C_, 0, st>>>(S.xnew, S.gcat, K1_, colOff);
  // ---- GATv2 layer 2 (K = C) ----
  cvt_bf16_k<<<cdiv(N2 * C_, TB), TB, 0, st>>>(S.xnew, S.bfA, N2 * C_);
  cvt_bf16_t_k<<<cdiv(C_ * C_, TB), TB, 0, st>>>(w2l, S.bfWl, C_, C_);
  cvt_bf16_t_k<<<cdiv(C_ * C_, TB), TB, 0, st>>>(w2r, S.bfWr, C_, C_);
  int tiles2 = (N2 / 16) * (C_ / 16);
  gemm_bf16_wmma_k<<<cdiv(tiles2, 8), gblk, 0, st>>>(S.bfA, S.bfWl, S.xl, N2, C_, C_);
  gemm_bf16_wmma_k<<<cdiv(tiles2, 8), gblk, 0, st>>>(S.bfA, S.bfWr, S.xr, N2, C_, C_);
  gat_tail(S.srcB, S.dstB, S.maskB, w2a, w2b, N2, S, st);
  // ---- pool 2: 256 -> 128 per graph (output reuses xl buffer) ----
  invnorm_k<<<1, C_, 0, st>>>(p2w, S.invn);
  node_score_k<<<cdiv(N2 * 32, TB), TB, 0, st>>>(S.xagg, p2w, S.invn, S.score, N2);
  float* xnew2 = S.xl;
  pool_k<<<B_, K1_, 0, st>>>(S.xagg, S.score, xnew2, S.newid, K1_, K2_);
  readout_k<<<B_, C_, 0, st>>>(xnew2, S.gcat, K2_, colOff);
}

extern "C" void kernel_launch(void* const* d_in, const int* in_sizes, int n_in,
                              void* d_out, int out_size, void* d_ws, size_t ws_size,
                              hipStream_t stream) {
  (void)in_sizes; (void)n_in; (void)out_size; (void)ws_size;
  const float* src_x = (const float*)d_in[0];
  const float* tgt_x = (const float*)d_in[1];
  const int*   src_ei = (const int*)d_in[2];
  const int*   tgt_ei = (const int*)d_in[3];
  const float* sw[10]; for (int i = 0; i < 10; ++i) sw[i] = (const float*)d_in[4 + i];
  const float* tw[10]; for (int i = 0; i < 10; ++i) tw[i] = (const float*)d_in[14 + i];
  const float* lin1_w = (const float*)d_in[24];
  const float* lin1_b = (const float*)d_in[25];
  const float* lin2_w = (const float*)d_in[26];
  const float* lin2_b = (const float*)d_in[27];
  const float* lin3_w = (const float*)d_in[28];
  const float* lin3_b = (const float*)d_in[29];

  // bump allocator over d_ws (~70 MB total)
  char* base = (char*)d_ws;
  size_t off = 0;
  auto alloc = [&](size_t bytes) -> void* {
    void* p = base + off;
    off += (bytes + 255) & ~(size_t)255;
    return p;
  };
  Scratch S;
  S.bfA   = (unsigned short*)alloc((size_t)2097152 * 2);      // max(N1*FIN, N2*C)
  S.bfWl  = (unsigned short*)alloc((size_t)C_ * C_ * 2);
  S.bfWr  = (unsigned short*)alloc((size_t)C_ * C_ * 2);
  S.xl    = (float*)alloc((size_t)N1 * C_ * 4);
  S.xr    = (float*)alloc((size_t)N1 * C_ * 4);
  S.xagg  = (float*)alloc((size_t)N1 * C_ * 4);
  S.xnew  = (float*)alloc((size_t)N2 * C_ * 4);
  S.esc   = (float*)alloc((size_t)(E_ + N1) * 4);
  S.emax  = (unsigned*)alloc((size_t)N1 * 4);
  S.denom = (float*)alloc((size_t)N1 * 4);
  S.score = (float*)alloc((size_t)N1 * 4);
  S.newid = (int*)alloc((size_t)N1 * 4);
  S.srcB  = (int*)alloc((size_t)E_ * 4);
  S.dstB  = (int*)alloc((size_t)E_ * 4);
  S.maskB = (int*)alloc((size_t)E_ * 4);
  S.invn  = (float*)alloc(256);
  S.gcat  = (float*)alloc((size_t)B_ * 512 * 4);
  S.h1    = (float*)alloc((size_t)B_ * 128 * 4);
  S.h2    = (float*)alloc((size_t)B_ * 64 * 4);
  S.h3    = (float*)alloc((size_t)B_ * 2 * 4);

  fill_f32_k<<<cdiv(B_ * 512, 256), 256, 0, stream>>>(S.gcat, 0.f, B_ * 512);

  run_branch(src_x, src_ei, src_ei + E_,
             sw[0], sw[1], sw[2], sw[3], sw[4], sw[5], sw[6], sw[7], sw[8], sw[9],
             0, S, stream);
  run_branch(tgt_x, tgt_ei, tgt_ei + E_,
             tw[0], tw[1], tw[2], tw[3], tw[4], tw[5], tw[6], tw[7], tw[8], tw[9],
             256, S, stream);

  linear_head_k<<<cdiv(64 * 128, 256), 256, 0, stream>>>(S.gcat, lin1_w, lin1_b, S.h1, 64, 512, 128, 1);
  linear_head_k<<<cdiv(64 * 64, 256), 256, 0, stream>>>(S.h1, lin2_w, lin2_b, S.h2, 64, 128, 64, 1);
  linear_head_k<<<1, 128, 0, stream>>>(S.h2, lin3_w, lin3_b, S.h3, 64, 64, 2, 0);
  logsoftmax2_k<<<1, 64, 0, stream>>>(S.h3, (float*)d_out, 64);
}
